// MambaModel_25761213841432
// MI455X (gfx1250) — compile-verified, matrix operand order checked by reading, not run on hardware
//
#include <hip/hip_runtime.h>
#include <hip/hip_bf16.h>
#include <math.h>

// ---------------- model dims ----------------
#define B_      8
#define SEQ_    1152
#define LEFF    1024
#define NCH     64
#define DMODEL  512
#define DINNER  1024
#define DSTATE  16
#define DTRANK  32
#define DCONV   4
#define NCLS    10
#define ROWS    (B_ * LEFF)        // 8192

// ---------------- WMMA types ----------------
typedef __attribute__((ext_vector_type(16))) __bf16 v16bf;
typedef __attribute__((ext_vector_type(8)))  float  v8f;

__device__ __forceinline__ unsigned int f2bf_u(float f) {
    unsigned int u = __float_as_uint(f);
    u += 0x7FFFu + ((u >> 16) & 1u);        // round-to-nearest-even
    return u >> 16;
}
__device__ __forceinline__ unsigned int pack2bf(float a, float b) {
    return f2bf_u(a) | (f2bf_u(b) << 16);
}

union BFrag {
    v16bf v;
    unsigned int w[8];
};

// ---------------- WMMA GEMM: C[M,N] = A[M,K] * B[N,K]^T (+bias) ----------------
// block = 256 threads = 8 waves; BM=256 (8 waves x 32 rows), BN=64, KT=32.
// Each wave: 2x4 grid of 16x16 WMMA tiles = 8 WMMAs per K-chunk.
// Double-buffered LDS, one barrier per chunk; all fragments preloaded so the
// WMMA chain issues without per-pair s_wait_dscnt stalls.
#define BM 256
#define BN 64
#define KT 32
#define PHW 36                      // LDS pitch (halfwords): mult of 4 (b64 stores), conflict-free frag gathers

// Per-lane fragment gather per CDNA5 16-bit 16x32 layout:
// lanes 0-15 hold row=lane; VGPR v<4 -> K = half*8 + 2v, v>=4 -> K = 16 + half*8 + 2(v-4)
__device__ __forceinline__ v16bf load_frag(const unsigned short* tile, int row, int half) {
    BFrag f;
    const unsigned short* p = tile + row * PHW + half * 8;
#pragma unroll
    for (int v = 0; v < 4; ++v) {
        f.w[v]     = *(const unsigned int*)(p + 2 * v);
        f.w[v + 4] = *(const unsigned int*)(p + 16 + 2 * v);
    }
    return f.v;
}

template<bool HAS_BIAS>
__global__ __launch_bounds__(256) void gemm_bt_bf16(
    const float* __restrict__ A, const float* __restrict__ B, float* __restrict__ C,
    int K, int lda, int ldb, int ldc, const float* __restrict__ bias)
{
    __shared__ unsigned short lds[2][(BM + BN) * PHW];

    const int tid  = threadIdx.x;
    const int bm   = blockIdx.y * BM;
    const int bn   = blockIdx.x * BN;
    const int lane = tid & 31;
    const int w    = tid >> 5;
    const int wm   = w * 32;          // 8 waves stacked along M; each covers all BN cols
    const int half = lane >> 4;
    const int lh   = lane & 15;

    v8f acc[2][4];
#pragma unroll
    for (int mi = 0; mi < 2; ++mi)
#pragma unroll
        for (int ni = 0; ni < 4; ++ni)
            acc[mi][ni] = (v8f)(0.0f);

    const int nch = K / KT;
    float4 ra[8], rb[2];

    // helpers ---------------------------------------------------------------
    auto fetch = [&](int c) {
        const int k0 = c * KT;
#pragma unroll
        for (int i = 0; i < 8; ++i) {                       // A: 256 rows x 8 float4
            int idx = tid + i * 256;
            int r = idx >> 3, q = idx & 7;
            ra[i] = *(const float4*)(A + (size_t)(bm + r) * lda + k0 + q * 4);
        }
#pragma unroll
        for (int i = 0; i < 2; ++i) {                       // B: 64 rows x 8 float4
            int idx = tid + i * 256;
            int r = idx >> 3, q = idx & 7;
            rb[i] = *(const float4*)(B + (size_t)(bn + r) * ldb + k0 + q * 4);
        }
    };
    auto stage = [&](unsigned short* buf) {
        unsigned short* As = buf;
        unsigned short* Bs = buf + BM * PHW;
#pragma unroll
        for (int i = 0; i < 8; ++i) {
            int idx = tid + i * 256;
            int r = idx >> 3, q = idx & 7;
            uint2 u; u.x = pack2bf(ra[i].x, ra[i].y); u.y = pack2bf(ra[i].z, ra[i].w);
            *(uint2*)(As + r * PHW + q * 4) = u;
        }
#pragma unroll
        for (int i = 0; i < 2; ++i) {
            int idx = tid + i * 256;
            int r = idx >> 3, q = idx & 7;
            uint2 u; u.x = pack2bf(rb[i].x, rb[i].y); u.y = pack2bf(rb[i].z, rb[i].w);
            *(uint2*)(Bs + r * PHW + q * 4) = u;
        }
    };
    auto compute = [&](const unsigned short* buf) {
        const unsigned short* As = buf;
        const unsigned short* Bs = buf + BM * PHW;
        // preload ALL fragments first so the WMMA chain can issue densely,
        // with the compiler retiring s_wait_dscnt progressively
        v16bf a0 = load_frag(As, wm + lh,      half);
        v16bf a1 = load_frag(As, wm + 16 + lh, half);
        v16bf b0 = load_frag(Bs, 0 * 16 + lh,  half);
        v16bf b1 = load_frag(Bs, 1 * 16 + lh,  half);
        v16bf b2 = load_frag(Bs, 2 * 16 + lh,  half);
        v16bf b3 = load_frag(Bs, 3 * 16 + lh,  half);
        acc[0][0] = __builtin_amdgcn_wmma_f32_16x16x32_bf16(false, a0, false, b0, (short)0, acc[0][0], false, false);
        acc[1][0] = __builtin_amdgcn_wmma_f32_16x16x32_bf16(false, a1, false, b0, (short)0, acc[1][0], false, false);
        acc[0][1] = __builtin_amdgcn_wmma_f32_16x16x32_bf16(false, a0, false, b1, (short)0, acc[0][1], false, false);
        acc[1][1] = __builtin_amdgcn_wmma_f32_16x16x32_bf16(false, a1, false, b1, (short)0, acc[1][1], false, false);
        acc[0][2] = __builtin_amdgcn_wmma_f32_16x16x32_bf16(false, a0, false, b2, (short)0, acc[0][2], false, false);
        acc[1][2] = __builtin_amdgcn_wmma_f32_16x16x32_bf16(false, a1, false, b2, (short)0, acc[1][2], false, false);
        acc[0][3] = __builtin_amdgcn_wmma_f32_16x16x32_bf16(false, a0, false, b3, (short)0, acc[0][3], false, false);
        acc[1][3] = __builtin_amdgcn_wmma_f32_16x16x32_bf16(false, a1, false, b3, (short)0, acc[1][3], false, false);
    };
    // -----------------------------------------------------------------------

    fetch(0);
    stage(lds[0]);
    __syncthreads();
    int p = 0;
    for (int c = 1; c < nch; ++c) {
        // prefetch chunk c+1 into near caches (global_prefetch_b8)
        if (c + 1 < nch)
            __builtin_prefetch(A + (size_t)(bm + tid) * lda + (c + 1) * KT, 0, 1);
        fetch(c);            // global -> regs
        stage(lds[p ^ 1]);   // regs -> other LDS buffer (overlaps with compute below)
        compute(lds[p]);     // WMMAs on current buffer
        __syncthreads();
        p ^= 1;
    }
    compute(lds[p]);

#pragma unroll
    for (int ni = 0; ni < 4; ++ni) {
        int col = bn + ni * 16 + lh;
        float bb = HAS_BIAS ? bias[col] : 0.0f;
#pragma unroll
        for (int mi = 0; mi < 2; ++mi) {
#pragma unroll
            for (int i = 0; i < 8; ++i) {
                int row = bm + wm + mi * 16 + half * 8 + i;
                C[(size_t)row * ldc + col] = acc[mi][ni][i] + bb;
            }
        }
    }
}

// ---------------- slice x[:, :1024, :] -> contiguous (8192,64) ----------------
__global__ __launch_bounds__(256) void slice_kernel(const float* __restrict__ x, float* __restrict__ xg) {
    int idx = blockIdx.x * 256 + threadIdx.x;              // 8192*64
    int c = idx & 63, row = idx >> 6;
    int b = row >> 10, t = row & 1023;
    xg[idx] = x[((size_t)b * SEQ_ + t) * NCH + c];
}

// ---------------- causal depthwise conv (k=4) + SiLU ----------------
__global__ __launch_bounds__(256) void conv_silu_kernel(
    const float* __restrict__ xz, const float* __restrict__ cw,
    const float* __restrict__ cb, float* __restrict__ xmc)
{
    int idx = blockIdx.x * 256 + threadIdx.x;              // 8192*1024
    int d = idx & 1023;
    int row = idx >> 10;
    int b = row >> 10, t = row & 1023;
    float acc = cb[d];
#pragma unroll
    for (int k = 0; k < DCONV; ++k) {
        int tt = t - (DCONV - 1) + k;
        if (tt >= 0)
            acc += xz[((size_t)(b * LEFF + tt)) * (2 * DINNER) + d] * cw[d * DCONV + k];
    }
    xmc[idx] = acc / (1.0f + __expf(-acc));
}

// ---------------- delta = softplus(dtraw + b_dt) in place ----------------
__global__ __launch_bounds__(256) void softplus_kernel(float* __restrict__ dta, const float* __restrict__ bdt) {
    int idx = blockIdx.x * 256 + threadIdx.x;
    int d = idx & 1023;
    float v = dta[idx] + bdt[d];
    dta[idx] = (v > 20.0f) ? v : log1pf(__expf(v));
}

// ---------------- selective scan: thread = (batch, channel), LDS-staged B/C ----------------
#define TS 64
__global__ __launch_bounds__(256) void scan_kernel(
    const float* __restrict__ xz, const float* __restrict__ xmc,
    const float* __restrict__ xdbc, const float* __restrict__ delta,
    const float* __restrict__ A_log, const float* __restrict__ Dv,
    float* __restrict__ yout)
{
    __shared__ float Bs[TS * DSTATE];
    __shared__ float Cs[TS * DSTATE];
    const int b = blockIdx.x >> 2;
    const int d = (blockIdx.x & 3) * 256 + threadIdx.x;

    float Arow[DSTATE];
#pragma unroll
    for (int n = 0; n < DSTATE; ++n) Arow[n] = -__expf(A_log[d * DSTATE + n]);
    const float Dd = Dv[d];

    float st[DSTATE];
#pragma unroll
    for (int n = 0; n < DSTATE; ++n) st[n] = 0.0f;

    for (int t0 = 0; t0 < LEFF; t0 += TS) {
        __syncthreads();
#pragma unroll
        for (int i = 0; i < 8; ++i) {                      // TS*32 floats, 8/thread
            int idx = threadIdx.x + i * 256;
            int s = idx >> 5, c = idx & 31;
            float v = xdbc[((size_t)(b * LEFF + t0 + s)) * 64 + DTRANK + c];
            if (c < DSTATE) Bs[s * DSTATE + c] = v;
            else            Cs[s * DSTATE + (c - DSTATE)] = v;
        }
        __syncthreads();

        for (int s = 0; s < TS; ++s) {
            size_t r = (size_t)(b * LEFF + t0 + s);
            float dt = delta[r * DINNER + d];
            float xv = xmc[r * DINNER + d];
            float dx = dt * xv;
            float accy = 0.0f;
#pragma unroll
            for (int n = 0; n < DSTATE; ++n) {
                float dA = __expf(dt * Arow[n]);
                st[n] = st[n] * dA + dx * Bs[s * DSTATE + n];
                accy += st[n] * Cs[s * DSTATE + n];
            }
            float y = accy + xv * Dd;
            float z = xz[r * (2 * DINNER) + DINNER + d];
            float sz = z / (1.0f + __expf(-z));
            yout[r * DINNER + d] = y * sz;
        }
    }
}

// ---------------- residual + LayerNorm (in place on h) ----------------
__global__ __launch_bounds__(256) void resln_kernel(
    const float* __restrict__ outl, float* __restrict__ h,
    const float* __restrict__ g, const float* __restrict__ bv)
{
    __shared__ float red[256];
    size_t r = blockIdx.x;
    int tid = threadIdx.x;
    float v0 = outl[r * DMODEL + tid]       + h[r * DMODEL + tid];
    float v1 = outl[r * DMODEL + 256 + tid] + h[r * DMODEL + 256 + tid];

    red[tid] = v0 + v1;
    __syncthreads();
    for (int s = 128; s > 0; s >>= 1) { if (tid < s) red[tid] += red[tid + s]; __syncthreads(); }
    float mean = red[0] * (1.0f / DMODEL);
    __syncthreads();

    red[tid] = v0 * v0 + v1 * v1;
    __syncthreads();
    for (int s = 128; s > 0; s >>= 1) { if (tid < s) red[tid] += red[tid + s]; __syncthreads(); }
    float var = red[0] * (1.0f / DMODEL) - mean * mean;
    float rstd = rsqrtf(var + 1e-5f);

    h[r * DMODEL + tid]       = (v0 - mean) * rstd * g[tid]       + bv[tid];
    h[r * DMODEL + 256 + tid] = (v1 - mean) * rstd * g[256 + tid] + bv[256 + tid];
}

// ---------------- mean pool over time ----------------
__global__ __launch_bounds__(256) void pool_kernel(const float* __restrict__ h, float* __restrict__ pooled) {
    int idx = blockIdx.x * 256 + threadIdx.x;              // 8*512
    int b = idx >> 9, c = idx & 511;
    float s = 0.0f;
    for (int t = 0; t < LEFF; ++t) s += h[((size_t)(b * LEFF + t)) * DMODEL + c];
    pooled[idx] = s * (1.0f / LEFF);
}

// ---------------- classifier head (8x10) ----------------
__global__ __launch_bounds__(128) void head_kernel(
    const float* __restrict__ pooled, const float* __restrict__ W,
    const float* __restrict__ bias, float* __restrict__ out)
{
    int tid = threadIdx.x;
    if (tid < B_ * NCLS) {
        int b = tid / NCLS, c = tid % NCLS;
        float s = bias[c];
        for (int dd = 0; dd < DMODEL; ++dd) s += pooled[b * DMODEL + dd] * W[c * DMODEL + dd];
        out[b * NCLS + c] = s;
    }
}

// ---------------- orchestration ----------------
extern "C" void kernel_launch(void* const* d_in, const int* in_sizes, int n_in,
                              void* d_out, int out_size, void* d_ws, size_t ws_size,
                              hipStream_t stream) {
    (void)in_sizes; (void)n_in; (void)out_size; (void)ws_size;

    const float* x        = (const float*)d_in[0];
    const float* W_lin_in = (const float*)d_in[1];
    const float* b_lin_in = (const float*)d_in[2];
    const float* W_lin_out= (const float*)d_in[3];
    const float* b_lin_out= (const float*)d_in[4];
    const float* W_in     = (const float*)d_in[5];
    const float* conv_w   = (const float*)d_in[6];
    const float* conv_b   = (const float*)d_in[7];
    const float* W_xproj  = (const float*)d_in[8];
    const float* W_dt     = (const float*)d_in[9];
    const float* b_dt     = (const float*)d_in[10];
    const float* A_log    = (const float*)d_in[11];
    const float* Dv       = (const float*)d_in[12];
    const float* W_out    = (const float*)d_in[13];
    const float* ln_g     = (const float*)d_in[14];
    const float* ln_b     = (const float*)d_in[15];
    float* out = (float*)d_out;

    float* ws   = (float*)d_ws;
    float* xg   = ws;                                   // 8192*64
    float* h    = xg   + (size_t)ROWS * NCH;            // 8192*512
    float* xz   = h    + (size_t)ROWS * DMODEL;         // 8192*2048
    float* xmc  = xz   + (size_t)ROWS * 2 * DINNER;     // 8192*1024
    float* xdbc = xmc  + (size_t)ROWS * DINNER;         // 8192*64
    float* dta  = xdbc + (size_t)ROWS * 64;             // 8192*1024
    float* yb   = dta  + (size_t)ROWS * DINNER;         // 8192*1024
    float* outl = yb   + (size_t)ROWS * DINNER;         // 8192*512
    float* pooled = outl + (size_t)ROWS * DMODEL;       // 8*512

    dim3 blk(256);

    // x slice -> contiguous
    slice_kernel<<<(ROWS * NCH) / 256, blk, 0, stream>>>(x, xg);

    // h = xg @ W_lin_in^T + b_lin_in   (8192,512,K=64)
    gemm_bt_bf16<true><<<dim3(DMODEL / BN, ROWS / BM), blk, 0, stream>>>(
        xg, W_lin_in, h, NCH, NCH, NCH, DMODEL, b_lin_in);

    for (int layer = 0; layer < 2; ++layer) {
        const float* Wi  = W_in   + (size_t)layer * 2 * DINNER * DMODEL;
        const float* cw  = conv_w + (size_t)layer * DINNER * DCONV;
        const float* cb  = conv_b + (size_t)layer * DINNER;
        const float* Wx  = W_xproj+ (size_t)layer * (DTRANK + 2 * DSTATE) * DINNER;
        const float* Wd  = W_dt   + (size_t)layer * DINNER * DTRANK;
        const float* bd  = b_dt   + (size_t)layer * DINNER;
        const float* Al  = A_log  + (size_t)layer * DINNER * DSTATE;
        const float* Dl  = Dv     + (size_t)layer * DINNER;
        const float* Wo  = W_out  + (size_t)layer * DMODEL * DINNER;
        const float* lg  = ln_g   + (size_t)layer * DMODEL;
        const float* lb  = ln_b   + (size_t)layer * DMODEL;

        // xz = h @ W_in^T   (8192,2048,K=512)
        gemm_bt_bf16<false><<<dim3((2 * DINNER) / BN, ROWS / BM), blk, 0, stream>>>(
            h, Wi, xz, DMODEL, DMODEL, DMODEL, 2 * DINNER, nullptr);

        // xm = silu(causal_dw_conv(xz[:, :1024]))
        conv_silu_kernel<<<(ROWS * DINNER) / 256, blk, 0, stream>>>(xz, cw, cb, xmc);

        // xdbc = xm @ W_xproj^T   (8192,64,K=1024)
        gemm_bt_bf16<false><<<dim3(64 / BN, ROWS / BM), blk, 0, stream>>>(
            xmc, Wx, xdbc, DINNER, DINNER, DINNER, 64, nullptr);

        // dtraw = xdbc[:, :32] @ W_dt^T   (8192,1024,K=32)
        gemm_bt_bf16<false><<<dim3(DINNER / BN, ROWS / BM), blk, 0, stream>>>(
            xdbc, Wd, dta, DTRANK, 64, DTRANK, DINNER, nullptr);

        // delta = softplus(dtraw + b_dt)
        softplus_kernel<<<(ROWS * DINNER) / 256, blk, 0, stream>>>(dta, bd);

        // selective scan (+ x*D, * silu(z))
        scan_kernel<<<B_ * (DINNER / 256), blk, 0, stream>>>(xz, xmc, xdbc, dta, Al, Dl, yb);

        // outl = y @ W_out^T   (8192,512,K=1024)
        gemm_bt_bf16<false><<<dim3(DMODEL / BN, ROWS / BM), blk, 0, stream>>>(
            yb, Wo, outl, DINNER, DINNER, DINNER, DMODEL, nullptr);

        // h = LayerNorm(outl + h)
        resln_kernel<<<ROWS, blk, 0, stream>>>(outl, h, lg, lb);
    }

    pool_kernel<<<(B_ * DMODEL) / 256, blk, 0, stream>>>(h, pooled);
    head_kernel<<<1, 128, 0, stream>>>(pooled, W_lin_out, b_lin_out, out);
}
